// ROTPooling_60619168416421
// MI455X (gfx1250) — matrix operand order repaired and checked by reading, not compiled, not run on hardware
//
#include <hip/hip_runtime.h>
#include <hip/hip_bf16.h>
#include <cstdint>
#include <cstddef>

#define BB     32
#define NN_    512
#define DD     256
#define NUMIT  4
#define EPS_   1e-8f
#define NSPLIT 8

typedef __attribute__((ext_vector_type(16))) __bf16        v16bf;
typedef __attribute__((ext_vector_type(8)))  float          v8f;
typedef __attribute__((ext_vector_type(8)))  unsigned short ushort8;
typedef __attribute__((ext_vector_type(4)))  unsigned int   u32x4;
typedef __attribute__((ext_vector_type(8)))  int            i32x8;
typedef __attribute__((ext_vector_type(4)))  int            i32x4;

union BfPack { v16bf v; ushort8 h[2]; };

// Toolchain hedge: therock-10.0 headers (6-arg tensor builtin) ship this header;
// ROCm 7.2 (5-arg builtin) does not.
#if __has_include(<hip/amd_detail/amd_gfx1250_TDM.h>)
#define TDM_6ARG 1
#else
#define TDM_6ARG 0
#endif

__device__ __forceinline__ unsigned short f2bf(float f) {
  union { float f; unsigned int u; } c; c.f = f;
  unsigned int u = c.u;
  u += 0x7fffu + ((u >> 16) & 1u);   // round-to-nearest-even
  return (unsigned short)(u >> 16);
}
__device__ __forceinline__ float softplus(float x) {
  return (x > 20.f) ? x : log1pf(expf(x));
}

// ---------------------------------------------------------------------------
// TDM: DMA one 2D bf16 tile (rows x 32 elements) from global into LDS,
// inserting 16B of padding after every 64B row => LDS stride 40 halves.
// D# encoding per CDNA5 ISA sec 8.3/8.4.
// ---------------------------------------------------------------------------
__device__ __forceinline__ void tdm_load_tile(unsigned lds_off,
                                              unsigned long long gaddr,
                                              int rows, int k_stride_elems) {
  u32x4 g0;
  g0[0] = 1u;                                         // count=1 (valid), user mode
  g0[1] = lds_off;                                    // lds_addr (bytes)
  g0[2] = (unsigned)(gaddr & 0xffffffffu);            // global_addr[31:0]
  g0[3] = ((unsigned)(gaddr >> 32) & 0x01ffffffu)     // global_addr[56:32]
          | 0x80000000u;                              // type=2 ("image")
  i32x8 g1;
  g1[0] = (1 << 16)        // data_size = 1 -> 2 bytes
        | (1 << 20)        // pad_enable
        | (3 << 22)        // pad_interval = 3 -> every 16 DWORDs (64B)
        | (3 << 25);       // pad_amount   = 3 -> 4 DWORDs (16B)
  g1[1] = (32u & 0xffffu) << 16;                      // tensor_dim0 = 32 (lo16)
  g1[2] = ((unsigned)rows & 0xffffu) << 16;           // dim0 hi=0 | tensor_dim1 lo
  g1[3] = ((unsigned)rows >> 16) | (32u << 16);       // tensor_dim1 hi | tile_dim0=32
  g1[4] = (unsigned)rows & 0xffffu;                   // tile_dim1=rows | tile_dim2=0
  g1[5] = (unsigned)k_stride_elems;                   // tensor_dim0_stride lo32
  g1[6] = 0; g1[7] = 0;                               // stride hi | dim1_stride=0
  i32x4 z4 = {0, 0, 0, 0};
#if TDM_6ARG
  i32x8 z8 = {0, 0, 0, 0, 0, 0, 0, 0};
  __builtin_amdgcn_tensor_load_to_lds(g0, g1, z4, z4, z8, 0);
#else
  __builtin_amdgcn_tensor_load_to_lds(g0, g1, z4, z4, 0);
#endif
}

// ---------------------------------------------------------------------------
// Batched NT GEMM, bf16 inputs (A:[b][M][K], Bt:[b][Ncol][K], both K-contig),
// fp32 WMMA accumulation. Block = 256 thr = 8 wave32, tile 128x64, K-tile 32.
// Double-buffered LDS staging via Tensor Data Mover (DMA for tile t+1 overlaps
// WMMA on tile t); emits v_wmma_f32_16x16x32_bf16.
// ---------------------------------------------------------------------------
template <int OBF>
__global__ __launch_bounds__(256)
void k_gemm_nt_bf16(const unsigned short* __restrict__ A,
                    const unsigned short* __restrict__ Bt,
                    void* __restrict__ Cout,
                    int M, int Ncol, int K,
                    size_t sA, size_t sB, size_t sC,
                    float cscale)
{
  __shared__ unsigned short lA[2][128 * 40];  // 128 rows x 32 (stride 40 = TDM pad)
  __shared__ unsigned short lB[2][64 * 40];   // 64 rows  x 32

  const int b   = blockIdx.z;
  const int m0  = blockIdx.x * 128;
  const int n0  = blockIdx.y * 64;
  const int tid = threadIdx.x;
  const int wave = tid >> 5;
  const int lane = tid & 31;
  const int wm = wave >> 1;      // 0..3 : 32-row strip
  const int wn = wave & 1;       // 0..1 : 32-col strip
  const int lm = lane & 15;
  const int lh = lane >> 4;

  const unsigned short* gA = A  + (size_t)b * sA + (size_t)m0 * K;
  const unsigned short* gB = Bt + (size_t)b * sB + (size_t)n0 * K;
  unsigned ldsA[2], ldsB[2];
  ldsA[0] = (unsigned)(unsigned long long)(&lA[0][0]);
  ldsA[1] = (unsigned)(unsigned long long)(&lA[1][0]);
  ldsB[0] = (unsigned)(unsigned long long)(&lB[0][0]);
  ldsB[1] = (unsigned)(unsigned long long)(&lB[1][0]);

  v8f acc[2][2] = {};
  const int nt = K >> 5;

  if (wave == 0) {                         // prime buffer 0 with tile 0
    tdm_load_tile(ldsA[0], (unsigned long long)(const void*)gA, 128, K);
    tdm_load_tile(ldsB[0], (unsigned long long)(const void*)gB, 64,  K);
  }

  for (int t = 0; t < nt; ++t) {
    const int cur = t & 1, nxt = cur ^ 1;
    __syncthreads();                       // buf[nxt] no longer being read
    if (wave == 0) {
      if (t + 1 < nt) {
        int kt = (t + 1) << 5;
        tdm_load_tile(ldsA[nxt], (unsigned long long)(const void*)(gA + kt), 128, K);
        tdm_load_tile(ldsB[nxt], (unsigned long long)(const void*)(gB + kt), 64,  K);
        __builtin_amdgcn_s_wait_tensorcnt(2);   // in-order: tile t's pair done
      } else {
        __builtin_amdgcn_s_wait_tensorcnt(0);
      }
    }
    __syncthreads();                       // buf[cur] visible to all waves

    const unsigned short* la = &lA[cur][0];
    const unsigned short* lb = &lB[cur][0];
    v16bf afrag[2], bfrag[2];
#pragma unroll
    for (int i = 0; i < 2; ++i) {
      int row = wm * 32 + i * 16 + lm;
      int ko  = lh * 8;                    // A layout: K in [ko..ko+7],[16+ko..16+ko+7]
      BfPack u;
      u.h[0] = *(const ushort8*)&la[row * 40 + ko];
      u.h[1] = *(const ushort8*)&la[row * 40 + 16 + ko];
      afrag[i] = u.v;
    }
#pragma unroll
    for (int j = 0; j < 2; ++j) {
      int row = wn * 32 + j * 16 + lm;     // Bt row = output column
      int kb  = lh * 8;                    // B layout: K in [kb..kb+7],[16+kb..16+kb+7]
      BfPack u;
      u.h[0] = *(const ushort8*)&lb[row * 40 + kb];
      u.h[1] = *(const ushort8*)&lb[row * 40 + 16 + kb];
      bfrag[j] = u.v;
    }
#pragma unroll
    for (int i = 0; i < 2; ++i)
#pragma unroll
      for (int j = 0; j < 2; ++j)
        acc[i][j] = __builtin_amdgcn_wmma_f32_16x16x32_bf16(
            false, afrag[i], false, bfrag[j], (short)0, acc[i][j], false, false);
  }

#pragma unroll
  for (int i = 0; i < 2; ++i) {
#pragma unroll
    for (int j = 0; j < 2; ++j) {
      int r0 = m0 + wm * 32 + i * 16 + lh * 8;   // C: VGPR v -> row r0+v, col c
      int c  = n0 + wn * 32 + j * 16 + lm;
#pragma unroll
      for (int v = 0; v < 8; ++v) {
        float val = acc[i][j][v] * cscale;
        size_t o = (size_t)b * sC + (size_t)(r0 + v) * Ncol + c;
        if (OBF) ((unsigned short*)Cout)[o] = f2bf(val);
        else     ((float*)Cout)[o] = val;
      }
    }
  }
}

// ---------------------------------------------------------------------------
// Elementwise / reduction kernels
// ---------------------------------------------------------------------------
__global__ void k_cvt_bf(const float* __restrict__ src,
                         unsigned short* __restrict__ dst, size_t n) {
  size_t gid = blockIdx.x * (size_t)blockDim.x + threadIdx.x;
  size_t stride = gridDim.x * (size_t)blockDim.x;
  for (size_t i = gid; i < n; i += stride) dst[i] = f2bf(src[i]);
}

// src f32 [b][R][C] -> dst bf16 [b][C][R], optional exp()
__global__ void k_trans_exp(const float* __restrict__ src,
                            unsigned short* __restrict__ dst,
                            int R, int C, int doExp) {
  __shared__ float t[32][33];
  int b = blockIdx.z;
  int c0 = blockIdx.x * 32, r0 = blockIdx.y * 32;
  int tx = threadIdx.x, ty = threadIdx.y;
  for (int j = ty; j < 32; j += 8) {
    float v = src[((size_t)b * R + (r0 + j)) * C + c0 + tx];
    t[j][tx] = doExp ? expf(v) : v;
  }
  __syncthreads();
  for (int j = ty; j < 32; j += 8)
    dst[((size_t)b * C + (c0 + j)) * R + r0 + tx] = f2bf(t[tx][j]);
}

__global__ void k_max_part(const float* __restrict__ src, size_t n,
                           float* __restrict__ part) {
  __shared__ float red[256];
  size_t gid = blockIdx.x * (size_t)blockDim.x + threadIdx.x;
  size_t stride = gridDim.x * (size_t)blockDim.x;
  float m = -3.4e38f;
  for (size_t i = gid; i < n; i += stride) m = fmaxf(m, src[i]);
  red[threadIdx.x] = m; __syncthreads();
  for (int s = 128; s > 0; s >>= 1) {
    if ((int)threadIdx.x < s) red[threadIdx.x] = fmaxf(red[threadIdx.x], red[threadIdx.x + s]);
    __syncthreads();
  }
  if (threadIdx.x == 0) part[blockIdx.x] = red[0];
}

__global__ void k_max_fin(const float* __restrict__ part, float* __restrict__ out) {
  __shared__ float red[256];
  red[threadIdx.x] = part[threadIdx.x]; __syncthreads();
  for (int s = 128; s > 0; s >>= 1) {
    if ((int)threadIdx.x < s) red[threadIdx.x] = fmaxf(red[threadIdx.x], red[threadIdx.x + s]);
    __syncthreads();
  }
  if (threadIdx.x == 0) *out = red[0];
}

__global__ void k_scale_bf(const float* __restrict__ src,
                           unsigned short* __restrict__ dst,
                           const float* __restrict__ maxp, size_t n) {
  float inv = 1.0f / (*maxp);
  size_t gid = blockIdx.x * (size_t)blockDim.x + threadIdx.x;
  size_t stride = gridDim.x * (size_t)blockDim.x;
  for (size_t i = gid; i < n; i += stride) dst[i] = f2bf(src[i] * inv);
}

__global__ void k_init(float* log_t, float* log_s, float* z,
                       float* mu, float* z1, float* eta, float* z2, size_t total) {
  const float Lq = logf(1.f / NN_ + EPS_);
  const float Lp = logf(1.f / DD + EPS_);
  size_t gid = blockIdx.x * (size_t)blockDim.x + threadIdx.x;
  size_t stride = gridDim.x * (size_t)blockDim.x;
  for (size_t i = gid; i < total; i += stride) {
    log_t[i] = Lq + Lp; log_s[i] = Lq + Lp; z[i] = 0.f;
  }
  if (gid < (size_t)BB * NN_) { mu[gid] = Lq; z1[gid] = 0.f; }
  if (gid < (size_t)BB * DD)  { eta[gid] = Lp; z2[gid] = 0.f; }
}

// T-update: softmax over feature dim (one block per (b,n) row, 256 thr = D)
__global__ __launch_bounds__(256)
void k_row_softmax_T(const float* __restrict__ x, const float* __restrict__ gw,
                     const float* __restrict__ z, const float* __restrict__ log_s,
                     const float* __restrict__ mu, float* __restrict__ log_t,
                     const float* a0p, const float* a1p, const float* rhop, int k) {
  __shared__ float red[256];
  const float a0 = softplus(a0p[k]), a1 = softplus(a1p[k]), rho = softplus(rhop[k]);
  const float inv = 1.f / (rho + a0);
  const size_t row = blockIdx.x;                 // b*N + n
  const int d = threadIdx.x;
  const size_t i = row * DD + d;
  float v = (x[i] + a1 * gw[i] + z[i] + rho * log_s[i]) * inv;
  red[d] = v; __syncthreads();
  for (int s = 128; s > 0; s >>= 1) { if (d < s) red[d] = fmaxf(red[d], red[d + s]); __syncthreads(); }
  float vmax = red[0]; __syncthreads();
  red[d] = expf(v - vmax); __syncthreads();
  for (int s = 128; s > 0; s >>= 1) { if (d < s) red[d] += red[d + s]; __syncthreads(); }
  float lse = vmax + logf(red[0]);
  log_t[i] = mu[row] + v - lse;
}

// S-update pass 1: y2 in-place over gw2, streaming logsumexp over n-split
__global__ __launch_bounds__(256)
void k_col_p1(float* __restrict__ gw, const float* __restrict__ z,
              const float* __restrict__ log_t, float* __restrict__ partials,
              const float* a1p, const float* rhop, int k) {
  const float a1 = softplus(a1p[k]), rho = softplus(rhop[k]);
  const float inv = 1.f / rho;
  int b = blockIdx.x, sp = blockIdx.y, d = threadIdx.x;
  int n0 = sp * (NN_ / NSPLIT);
  float m = -3.4e38f, s = 0.f;
  for (int n = n0; n < n0 + NN_ / NSPLIT; ++n) {
    size_t i = ((size_t)(b * NN_ + n)) * DD + d;
    float y2 = (a1 * gw[i] - z[i] + rho * log_t[i]) * inv;   // neg==0 (full mask)
    gw[i] = y2;
    if (y2 > m) { s = s * expf(m - y2) + 1.f; m = y2; } else s += expf(y2 - m);
  }
  size_t p = ((size_t)(b * NSPLIT + sp) * DD + d) * 2;
  partials[p] = m; partials[p + 1] = s;
}

__global__ void k_col_merge(const float* __restrict__ partials, float* __restrict__ lse) {
  int b = blockIdx.x, d = threadIdx.x;
  float m = -3.4e38f, s = 0.f;
  for (int sp = 0; sp < NSPLIT; ++sp) {
    size_t p = ((size_t)(b * NSPLIT + sp) * DD + d) * 2;
    float m2 = partials[p], s2 = partials[p + 1];
    if (m2 > m) { s = s * expf(m - m2) + s2; m = m2; }
    else        { s += s2 * expf(m2 - m); }
  }
  lse[b * DD + d] = m + logf(s);
}

__global__ void k_col_p2(const float* __restrict__ y2, const float* __restrict__ eta,
                         const float* __restrict__ lse, float* __restrict__ log_s) {
  int b = blockIdx.x, sp = blockIdx.y, d = threadIdx.x;
  float e = eta[b * DD + d], l = lse[b * DD + d];
  int n0 = sp * (NN_ / NSPLIT);
  for (int n = n0; n < n0 + NN_ / NSPLIT; ++n) {
    size_t i = ((size_t)(b * NN_ + n)) * DD + d;
    log_s[i] = e + y2[i] - l;
  }
}

// marginal prox + duals; uses logsumexp identities: log_t1==mu_old, log_s1==eta_old
__global__ void k_marg(float* mu, float* z1, float* eta, float* z2,
                       const float* a2p, const float* a3p, const float* rhop, int k) {
  const float a2 = softplus(a2p[k]), a3 = softplus(a3p[k]), rho = softplus(rhop[k]);
  const float Lq = logf(1.f / NN_ + EPS_), Lp = logf(1.f / DD + EPS_);
  int gid = blockIdx.x * blockDim.x + threadIdx.x;
  if (gid < BB * NN_) {
    float old = mu[gid];
    float nm = (a2 * Lq + rho * old - z1[gid]) / (a2 + rho);
    z1[gid] += rho * (expf(nm) - expf(old));
    mu[gid] = nm;
  }
  if (gid < BB * DD) {
    float old = eta[gid];
    float ne = (a3 * Lp + rho * old - z2[gid]) / (a3 + rho);
    z2[gid] += rho * (expf(ne) - expf(old));
    eta[gid] = ne;
  }
}

__global__ void k_zupd(float* __restrict__ z, const float* __restrict__ log_t,
                       const float* __restrict__ log_s, const float* rhop, int k, size_t n) {
  const float rho = softplus(rhop[k]);
  size_t gid = blockIdx.x * (size_t)blockDim.x + threadIdx.x;
  size_t stride = gridDim.x * (size_t)blockDim.x;
  for (size_t i = gid; i < n; i += stride)
    z[i] += rho * (expf(log_t[i]) - expf(log_s[i]));
}

__global__ void k_zero(float* p, size_t n) {
  size_t gid = blockIdx.x * (size_t)blockDim.x + threadIdx.x;
  if (gid < n) p[gid] = 0.f;
}

__global__ void k_final(const float* __restrict__ x, const float* __restrict__ log_t,
                        float* __restrict__ out) {
  int b = blockIdx.x, sp = blockIdx.y, d = threadIdx.x;
  float acc = 0.f;
  int n0 = sp * (NN_ / NSPLIT);
  for (int n = n0; n < n0 + NN_ / NSPLIT; ++n) {
    size_t i = ((size_t)(b * NN_ + n)) * DD + d;
    acc += x[i] * expf(log_t[i]);
  }
  atomicAdd(&out[b * DD + d], (float)DD * acc);
}

// ---------------------------------------------------------------------------
extern "C" void kernel_launch(void* const* d_in, const int* in_sizes, int n_in,
                              void* d_out, int out_size, void* d_ws, size_t ws_size,
                              hipStream_t stream) {
  (void)in_sizes; (void)n_in; (void)out_size; (void)ws_size;
  const float* x    = (const float*)d_in[0];
  const float* a0p  = (const float*)d_in[1];
  const float* a1p  = (const float*)d_in[2];
  const float* a2p  = (const float*)d_in[3];
  const float* a3p  = (const float*)d_in[4];
  const float* rhop = (const float*)d_in[5];
  float* out = (float*)d_out;

  char* ws = (char*)d_ws;
  size_t o = 0;
  auto alloc = [&](size_t bytes) -> void* {
    void* p = ws + o; o = (o + bytes + 255) & ~(size_t)255; return p;
  };

  const size_t nBND = (size_t)BB * NN_ * DD;           // 4.2M
  float*          c2f   = (float*)alloc((size_t)BB * NN_ * NN_ * 4);   // 32MB (reused below)
  float*          c1f   = (float*)alloc((size_t)BB * DD * DD * 4);     // 8MB  (reused below)
  unsigned short* c1bf  = (unsigned short*)alloc((size_t)BB * DD * DD * 2);
  unsigned short* c2bf  = (unsigned short*)alloc((size_t)BB * NN_ * NN_ * 2);
  unsigned short* xbf   = (unsigned short*)alloc(nBND * 2);
  unsigned short* xbfT  = (unsigned short*)alloc(nBND * 2);
  float*          log_t = (float*)alloc(nBND * 4);
  float*          log_s = (float*)alloc(nBND * 4);
  float*          zbuf  = (float*)alloc(nBND * 4);
  float*          mu    = (float*)alloc((size_t)BB * NN_ * 4);
  float*          z1    = (float*)alloc((size_t)BB * NN_ * 4);
  float*          eta   = (float*)alloc((size_t)BB * DD * 4);
  float*          z2    = (float*)alloc((size_t)BB * DD * 4);
  float*          maxbuf= (float*)alloc(4096);
  // Aliases into dead-after-setup regions:
  float*          gw    = (float*)c2f;                                // 16MB
  unsigned short* t1bf  = (unsigned short*)((char*)c2f + (16u << 20)); // 8MB
  unsigned short* ebfT  = (unsigned short*)((char*)c2f + (24u << 20)); // 8MB
  float*          partials = (float*)c1f;                             // 512KB
  float*          lse   = (float*)((char*)c1f + (1u << 20));          // 32KB

  float* part1 = maxbuf;
  float* part2 = maxbuf + 256;
  float* smax1 = maxbuf + 512;
  float* smax2 = maxbuf + 513;

  auto gemm = [&](const unsigned short* A, const unsigned short* Bt, void* C,
                  int M, int Ncol, int K, size_t sA, size_t sB, size_t sC,
                  float scale, int obf) {
    dim3 g(M / 128, Ncol / 64, BB);
    if (obf) k_gemm_nt_bf16<1><<<g, dim3(256), 0, stream>>>(A, Bt, C, M, Ncol, K, sA, sB, sC, scale);
    else     k_gemm_nt_bf16<0><<<g, dim3(256), 0, stream>>>(A, Bt, C, M, Ncol, K, sA, sB, sC, scale);
  };
  const dim3 tb(32, 8);
  const dim3 tg(DD / 32, NN_ / 32, BB);

  // ---- setup: bf16 copies, Gram matrices, max-normalize ----
  k_cvt_bf<<<1024, 256, 0, stream>>>(x, xbf, nBND);
  k_trans_exp<<<tg, tb, 0, stream>>>(x, xbfT, NN_, DD, 0);
  gemm(xbf,  xbf,  c2f, NN_, NN_, DD,  (size_t)NN_ * DD, (size_t)NN_ * DD, (size_t)NN_ * NN_, 1.f / DD, 0);
  gemm(xbfT, xbfT, c1f, DD,  DD,  NN_, (size_t)DD * NN_, (size_t)DD * NN_, (size_t)DD * DD,  1.f / NN_, 0);
  k_max_part<<<256, 256, 0, stream>>>(c1f, (size_t)BB * DD * DD, part1);
  k_max_fin<<<1, 256, 0, stream>>>(part1, smax1);
  k_max_part<<<256, 256, 0, stream>>>(c2f, (size_t)BB * NN_ * NN_, part2);
  k_max_fin<<<1, 256, 0, stream>>>(part2, smax2);
  k_scale_bf<<<1024, 256, 0, stream>>>(c1f, c1bf, smax1, (size_t)BB * DD * DD);
  k_scale_bf<<<1024, 256, 0, stream>>>(c2f, c2bf, smax2, (size_t)BB * NN_ * NN_);
  k_init<<<1024, 256, 0, stream>>>(log_t, log_s, zbuf, mu, z1, eta, z2, nBND);

  // ---- Bregman ADMM iterations ----
  for (int k = 0; k < NUMIT; ++k) {
    // gw = c2 * exp(S) * c1   (a1 applied in the softmax kernel)
    k_trans_exp<<<tg, tb, 0, stream>>>(log_s, ebfT, NN_, DD, 1);
    gemm(c2bf, ebfT, t1bf, NN_, DD, NN_, (size_t)NN_ * NN_, (size_t)DD * NN_, (size_t)NN_ * DD, 1.f, 1);
    gemm(t1bf, c1bf, gw,   NN_, DD, DD,  (size_t)NN_ * DD,  (size_t)DD * DD,  (size_t)NN_ * DD, 1.f, 0);
    k_row_softmax_T<<<BB * NN_, 256, 0, stream>>>(x, gw, zbuf, log_s, mu, log_t, a0p, a1p, rhop, k);

    // gw2 = c2 * exp(T) * c1
    k_trans_exp<<<tg, tb, 0, stream>>>(log_t, ebfT, NN_, DD, 1);
    gemm(c2bf, ebfT, t1bf, NN_, DD, NN_, (size_t)NN_ * NN_, (size_t)DD * NN_, (size_t)NN_ * DD, 1.f, 1);
    gemm(t1bf, c1bf, gw,   NN_, DD, DD,  (size_t)NN_ * DD,  (size_t)DD * DD,  (size_t)NN_ * DD, 1.f, 0);
    k_col_p1<<<dim3(BB, NSPLIT), 256, 0, stream>>>(gw, zbuf, log_t, partials, a1p, rhop, k);
    k_col_merge<<<BB, 256, 0, stream>>>(partials, lse);
    k_col_p2<<<dim3(BB, NSPLIT), 256, 0, stream>>>(gw, eta, lse, log_s);

    k_marg<<<64, 256, 0, stream>>>(mu, z1, eta, z2, a2p, a3p, rhop, k);
    k_zupd<<<1024, 256, 0, stream>>>(zbuf, log_t, log_s, rhop, k, nBND);
  }

  // ---- out[b,d] = D * sum_n x * exp(log_t) ----
  k_zero<<<(BB * DD + 255) / 256, 256, 0, stream>>>(out, (size_t)BB * DD);
  k_final<<<dim3(BB, NSPLIT), 256, 0, stream>>>(x, log_t, out);
}
